// SwinMemAE_39204461478041
// MI455X (gfx1250) — matrix operand order repaired and checked by reading, not compiled, not run on hardware
//
#include <hip/hip_runtime.h>
#include <hip/hip_bf16.h>

#define BATCH 4096
#define SEQ 49
#define DIM 32
#define MEMSZ 4096
#define MEMD 256
#define TOPKN 10

typedef __attribute__((ext_vector_type(16))) __bf16 v16bf;
typedef __attribute__((ext_vector_type(8)))  float  v8f;

#if defined(__has_builtin)
#if __has_builtin(__builtin_amdgcn_tensor_load_to_lds)
#define HAVE_TDM 1
#endif
#endif

#if defined(HAVE_TDM)
#if __has_include(<hip/amd_detail/amd_gfx1250_TDM.h>)
#define TDM_6ARG 1
#endif
typedef unsigned int u32x4 __attribute__((ext_vector_type(4)));
typedef int          i32x4 __attribute__((ext_vector_type(4)));
typedef int          i32x8 __attribute__((ext_vector_type(8)));
#endif

// LDS tile pitch: 36 dwords = 144B (multiple of 16 for B128 async writes;
// gcd(36,64)=4 -> 16 consecutive rows map to 16 distinct banks).
#define TPITCH 36

// ---------------------------------------------------------------------------
// WMMA GEMM:  C[M,N] = A[M,K] @ W[N,K]^T (+ bias[n])
// Workgroup (256 thr, 8 waves) computes a 32x128 macro-tile.
// Wave w: mw = w>>2 selects 16-row half, nw = w&3 selects 32-col strip
// -> 2 WMMAs per wave per K-step (K-chunk = 32), A-fragment shared.
// A-slab (32x32 f32) staged via GLOBAL_LOAD_ASYNC_TO_LDS_B128 (ASYNCcnt);
// W-slab (128x32 f32) staged via the Tensor Data Mover (TENSORcnt) with
// LDS padding producing the 36-dword pitch; async fallback if no TDM builtin.
// ---------------------------------------------------------------------------
__global__ void __launch_bounds__(256)
gemm_bf16_wmma(const float* __restrict__ A, const float* __restrict__ W,
               const float* __restrict__ bias, float* __restrict__ C,
               int M, int N, int K) {
  __shared__ __align__(16) float shA[32 * TPITCH];     //  4.6 KB
  __shared__ __align__(16) float shW[128 * TPITCH];    // 18.4 KB

  const int tid  = threadIdx.x;
  const int lane = tid & 31;
  const int wv   = tid >> 5;          // 0..7
  const int mw   = wv >> 2;           // 0..1
  const int nw   = wv & 3;            // 0..3

  const int nblk = (N + 127) >> 7;    // macro-tiles along N
  const int m0 = (blockIdx.x / nblk) * 32;
  const int n0 = (blockIdx.x % nblk) * 128;

  const int half = lane >> 4;
  const int l16  = lane & 15;

  // A-staging assignment: thread t -> row t/8 (0..31), 16B quad t%8
  const int arow  = tid >> 3;
  const int aquad = tid & 7;
  const uint32_t shA_base = (uint32_t)(uintptr_t)&shA[0];
  const uint32_t shW_base = (uint32_t)(uintptr_t)&shW[0];
  const uint32_t ldsA = shA_base + (uint32_t)(arow * TPITCH * 4 + aquad * 16);

  v8f acc[2];
  acc[0] = (v8f){};
  acc[1] = (v8f){};

  for (int k0 = 0; k0 < K; k0 += 32) {
    // ---- stage A (32x32 f32) : async copy global -> LDS ----
    {
      const uint64_t ga = (uint64_t)(A + (size_t)(m0 + arow) * K + k0 + aquad * 4);
      asm volatile("global_load_async_to_lds_b128 %0, %1, off"
                   :: "v"(ldsA), "v"(ga) : "memory");
    }

    // ---- stage W (128x32 f32) ----
#if defined(HAVE_TDM)
    if (wv == 0) {
      const uint64_t ga = (uint64_t)(W + (size_t)n0 * K + k0);
      const uint32_t rows = ((N - n0) < 128) ? (uint32_t)(N - n0) : 128u;
      u32x4 g0;
      g0[0] = 1u;                                        // count=1 (valid user D#)
      g0[1] = shW_base;                                  // lds_addr
      g0[2] = (uint32_t)(ga & 0xffffffffu);              // global_addr[31:0]
      g0[3] = (uint32_t)((ga >> 32) & 0x01ffffffu) | 0x80000000u; // [56:32] | type=2
      i32x8 g1;
      g1[0] = (int)((2u << 16)        // data_size = 4B
                  | (1u << 20)        // pad_enable
                  | (4u << 22)        // pad_interval: every 32 dwords
                  | (3u << 25));      // pad_amount: 4 dwords -> pitch 36
      g1[1] = (int)(((uint32_t)K & 0xffffu) << 16);                       // tensor_dim0 lo
      g1[2] = (int)((((uint32_t)K >> 16) & 0xffffu) | ((rows & 0xffffu) << 16)); // td0 hi|td1 lo
      g1[3] = (int)(32u << 16);                                           // td1 hi=0 | tile_dim0=32
      g1[4] = (int)128u;                                                  // tile_dim1=128, tile_dim2=0
      g1[5] = (int)K;                                                     // tensor_dim0_stride lo32
      g1[6] = 0;                                                          // stride0 hi | stride1 lo
      g1[7] = 0;
      i32x4 z4 = {0, 0, 0, 0};
#if defined(TDM_6ARG)
      i32x8 z8 = {0, 0, 0, 0, 0, 0, 0, 0};
      __builtin_amdgcn_tensor_load_to_lds(g0, g1, z4, z4, z8, 0);
#else
      __builtin_amdgcn_tensor_load_to_lds(g0, g1, z4, z4, 0);
#endif
    }
    if (wv == 0) __builtin_amdgcn_s_wait_tensorcnt(0);
#else
#pragma unroll
    for (int c = 0; c < 4; ++c) {
      const int idx = tid + c * 256;          // 0..1023
      const int wr = idx >> 3, wq = idx & 7;  // row 0..127, quad 0..7
      const int wrg = (n0 + wr < N) ? (n0 + wr) : (N - 1);
      const uint32_t ldsW = shW_base + (uint32_t)(wr * TPITCH * 4 + wq * 16);
      const uint64_t ga = (uint64_t)(W + (size_t)wrg * K + k0 + wq * 4);
      asm volatile("global_load_async_to_lds_b128 %0, %1, off"
                   :: "v"(ldsW), "v"(ga) : "memory");
    }
#endif
    asm volatile("s_wait_asynccnt 0x0" ::: "memory");
    __syncthreads();

    // ---- build fragments from LDS (f32 -> bf16) and issue 2 WMMAs ----
    v16bf a, b0, b1;
    {
      const float* ar = shA + (mw * 16 + l16) * TPITCH;
#pragma unroll
      for (int j = 0; j < 8; ++j) {
        a[j]     = (__bf16)ar[half * 8 + j];
        a[j + 8] = (__bf16)ar[16 + half * 8 + j];
      }
    }
    {
      const float* w0 = shW + (nw * 32 + l16) * TPITCH;
      const float* w1 = shW + (nw * 32 + 16 + l16) * TPITCH;
#pragma unroll
      for (int j = 0; j < 8; ++j) {
        b0[j]     = (__bf16)w0[half * 8 + j];
        b0[j + 8] = (__bf16)w0[16 + half * 8 + j];
        b1[j]     = (__bf16)w1[half * 8 + j];
        b1[j + 8] = (__bf16)w1[16 + half * 8 + j];
      }
    }
    acc[0] = __builtin_amdgcn_wmma_f32_16x16x32_bf16(false, a, false, b0,
                                                     (short)0, acc[0], false, false);
    acc[1] = __builtin_amdgcn_wmma_f32_16x16x32_bf16(false, a, false, b1,
                                                     (short)0, acc[1], false, false);
    __syncthreads();   // protect LDS before restaging
  }

  // ---- store: C/D layout -> lane holds col l16, VGPR r -> row r + 8*half ----
  const int mbase = m0 + mw * 16;
#pragma unroll
  for (int sub = 0; sub < 2; ++sub) {
    const int nsub = n0 + nw * 32 + sub * 16;
    if (nsub >= N) continue;                   // wave-uniform guard
    const int n = nsub + l16;
    const float bn = bias ? bias[n] : 0.0f;
#pragma unroll
    for (int r = 0; r < 8; ++r) {
      const int row = mbase + r + 8 * half;
      C[(size_t)row * N + n] = acc[sub][r] + bn;
    }
  }
}

// ---------------------------------------------------------------------------
// Fused per-image encoder: patch conv + 2 transformer blocks, all in LDS.
// One workgroup (256 threads) per image.
// ---------------------------------------------------------------------------
__global__ void swin_encoder_kernel(
    const float* __restrict__ x,        // [B,1,28,28]
    const float* __restrict__ patch_w,  // [32,1,4,4]
    const float* __restrict__ patch_b,  // [32]
    const float* __restrict__ ln1_g, const float* __restrict__ ln1_b,   // [2,32]
    const float* __restrict__ inW,  const float* __restrict__ inB,      // [2,96,32],[2,96]
    const float* __restrict__ outW, const float* __restrict__ outB,     // [2,32,32],[2,32]
    const float* __restrict__ ln2_g, const float* __restrict__ ln2_b,   // [2,32]
    const float* __restrict__ f1W, const float* __restrict__ f1B,       // [2,128,32],[2,128]
    const float* __restrict__ f2W, const float* __restrict__ f2B,       // [2,32,128],[2,32]
    float* __restrict__ t_out)          // [B,1568]
{
  __shared__ float t[SEQ][DIM];       // tokens (residual stream)
  __shared__ float ln[SEQ][DIM];      // layernorm scratch
  __shared__ float tmp[SEQ][DIM];     // attention output
  __shared__ float big[9506];         // union: qkv[49*96] + att[2*49*49] | ff[49*128]
  float* qkv = big;                   // [49][96]
  float* att = big + SEQ * 96;        // [2][49][49]
  float* ff  = big;                   // [49][128] (after attention phase)

  const int b   = blockIdx.x;
  const int tid = threadIdx.x;
  const int nt  = blockDim.x;
  const float* xb = x + (size_t)b * 784;

  // ---- patch conv 4x4 stride 4 : t[s][o] ----
  for (int i = tid; i < SEQ * DIM; i += nt) {
    const int s = i / DIM, o = i % DIM;
    const int py = s / 7, px = s % 7;
    float acc = patch_b[o];
#pragma unroll
    for (int ky = 0; ky < 4; ++ky)
#pragma unroll
      for (int kx = 0; kx < 4; ++kx)
        acc += xb[(4 * py + ky) * 28 + 4 * px + kx] * patch_w[o * 16 + ky * 4 + kx];
    t[s][o] = acc;
  }
  __syncthreads();

  for (int blk = 0; blk < 2; ++blk) {
    const float* g1 = ln1_g + blk * 32; const float* b1 = ln1_b + blk * 32;
    const float* Wi = inW  + blk * 96 * 32;  const float* Bi = inB  + blk * 96;
    const float* Wo = outW + blk * 32 * 32;  const float* Bo = outB + blk * 32;
    const float* g2 = ln2_g + blk * 32; const float* b2 = ln2_b + blk * 32;
    const float* W1 = f1W + blk * 128 * 32;  const float* B1 = f1B + blk * 128;
    const float* W2 = f2W + blk * 32 * 128;  const float* B2 = f2B + blk * 32;

    // LN1
    for (int s = tid; s < SEQ; s += nt) {
      float mu = 0.f;
      for (int d = 0; d < DIM; ++d) mu += t[s][d];
      mu *= (1.f / DIM);
      float var = 0.f;
      for (int d = 0; d < DIM; ++d) { float z = t[s][d] - mu; var += z * z; }
      var *= (1.f / DIM);
      const float inv = rsqrtf(var + 1e-5f);
      for (int d = 0; d < DIM; ++d) ln[s][d] = (t[s][d] - mu) * inv * g1[d] + b1[d];
    }
    __syncthreads();

    // qkv projection: qkv[s][r], r in [0,96)
    for (int i = tid; i < SEQ * 96; i += nt) {
      const int s = i / 96, r = i % 96;
      float acc = Bi[r];
      for (int d = 0; d < DIM; ++d) acc += ln[s][d] * Wi[r * 32 + d];
      qkv[s * 96 + r] = acc;
    }
    __syncthreads();

    // scores: att[h][s][u] = (q_s . k_u) / 4
    for (int i = tid; i < 2 * SEQ * SEQ; i += nt) {
      const int h = i / (SEQ * SEQ);
      const int rem = i % (SEQ * SEQ);
      const int s = rem / SEQ, u = rem % SEQ;
      float acc = 0.f;
      for (int d = 0; d < 16; ++d)
        acc += qkv[s * 96 + h * 16 + d] * qkv[u * 96 + 32 + h * 16 + d];
      att[i] = acc * 0.25f;
    }
    __syncthreads();

    // softmax over last dim (rows of 49)
    for (int r = tid; r < 2 * SEQ; r += nt) {
      float* row = att + r * SEQ;
      float mx = row[0];
      for (int u = 1; u < SEQ; ++u) mx = fmaxf(mx, row[u]);
      float sum = 0.f;
      for (int u = 0; u < SEQ; ++u) { float e = __expf(row[u] - mx); row[u] = e; sum += e; }
      const float inv = 1.f / sum;
      for (int u = 0; u < SEQ; ++u) row[u] *= inv;
    }
    __syncthreads();

    // attention output: tmp[s][h*16+d] = sum_u att[h][s][u] * v[u][h*16+d]
    for (int i = tid; i < SEQ * DIM; i += nt) {
      const int s = i / DIM, e = i % DIM;
      const int h = e / 16, d = e % 16;
      const float* arow = att + (h * SEQ + s) * SEQ;
      float acc = 0.f;
      for (int u = 0; u < SEQ; ++u) acc += arow[u] * qkv[u * 96 + 64 + h * 16 + d];
      tmp[s][e] = acc;
    }
    __syncthreads();

    // out-proj + residual into t
    for (int i = tid; i < SEQ * DIM; i += nt) {
      const int s = i / DIM, d = i % DIM;
      float acc = Bo[d];
      for (int e = 0; e < DIM; ++e) acc += tmp[s][e] * Wo[d * 32 + e];
      t[s][d] = t[s][d] + acc;
    }
    __syncthreads();

    // LN2
    for (int s = tid; s < SEQ; s += nt) {
      float mu = 0.f;
      for (int d = 0; d < DIM; ++d) mu += t[s][d];
      mu *= (1.f / DIM);
      float var = 0.f;
      for (int d = 0; d < DIM; ++d) { float z = t[s][d] - mu; var += z * z; }
      var *= (1.f / DIM);
      const float inv = rsqrtf(var + 1e-5f);
      for (int d = 0; d < DIM; ++d) ln[s][d] = (t[s][d] - mu) * inv * g2[d] + b2[d];
    }
    __syncthreads();

    // ff1 + exact gelu -> ff[s][r] (overlays qkv/att region; both dead now)
    for (int i = tid; i < SEQ * 128; i += nt) {
      const int s = i / 128, r = i % 128;
      float acc = B1[r];
      for (int d = 0; d < DIM; ++d) acc += ln[s][d] * W1[r * 32 + d];
      ff[i] = 0.5f * acc * (1.f + erff(acc * 0.70710678118654752f));
    }
    __syncthreads();

    // ff2 + residual
    for (int i = tid; i < SEQ * DIM; i += nt) {
      const int s = i / DIM, d = i % DIM;
      float acc = B2[d];
      for (int r = 0; r < 128; ++r) acc += ff[s * 128 + r] * W2[d * 128 + r];
      t[s][d] = t[s][d] + acc;
    }
    __syncthreads();
  }

  // flatten [49,32] row-major -> t_out[b][1568]
  for (int i = tid; i < SEQ * DIM; i += nt)
    t_out[(size_t)b * (SEQ * DIM) + i] = t[i / DIM][i % DIM];
}

// ---------------------------------------------------------------------------
// Row L2-normalize: out[r,:] = in[r,:] / max(||in[r,:]||, 1e-12)
// ---------------------------------------------------------------------------
__global__ void row_normalize(const float* __restrict__ in,
                              float* __restrict__ out, int cols) {
  __shared__ float red[256];
  const int row = blockIdx.x;
  const float* r = in + (size_t)row * cols;
  float s = 0.f;
  for (int c = threadIdx.x; c < cols; c += blockDim.x) { const float v = r[c]; s += v * v; }
  red[threadIdx.x] = s;
  __syncthreads();
  for (int off = 128; off > 0; off >>= 1) {
    if (threadIdx.x < off) red[threadIdx.x] += red[threadIdx.x + off];
    __syncthreads();
  }
  const float inv = 1.f / fmaxf(sqrtf(red[0]), 1e-12f);
  for (int c = threadIdx.x; c < cols; c += blockDim.x)
    out[(size_t)row * cols + c] = r[c] * inv;
}

// ---------------------------------------------------------------------------
// Top-10 over sim row + softmax + weighted gather of ORIGINAL memory rows.
// ---------------------------------------------------------------------------
__global__ void topk_mem_read(const float* __restrict__ sim,    // [B, MEMSZ]
                              const float* __restrict__ memory, // [MEMSZ, MEMD]
                              float* __restrict__ mem_lat) {    // [B, MEMD]
  __shared__ float bval[256];
  __shared__ int   bidx[256];
  __shared__ int   chosen[TOPKN];
  __shared__ float cval[TOPKN];
  __shared__ float w[TOPKN];
  const int row = blockIdx.x;
  const float* s = sim + (size_t)row * MEMSZ;

  for (int k = 0; k < TOPKN; ++k) {
    float best = -3.402823e38f;
    int besti = 0x7fffffff;
    for (int c = threadIdx.x; c < MEMSZ; c += blockDim.x) {
      bool taken = false;
      for (int j = 0; j < k; ++j) taken = taken || (chosen[j] == c);
      if (taken) continue;
      const float v = s[c];
      if (v > best || (v == best && c < besti)) { best = v; besti = c; }
    }
    bval[threadIdx.x] = best;
    bidx[threadIdx.x] = besti;
    __syncthreads();
    for (int off = 128; off > 0; off >>= 1) {
      if (threadIdx.x < off) {
        const float v2 = bval[threadIdx.x + off];
        const int   i2 = bidx[threadIdx.x + off];
        if (v2 > bval[threadIdx.x] ||
            (v2 == bval[threadIdx.x] && i2 < bidx[threadIdx.x])) {
          bval[threadIdx.x] = v2; bidx[threadIdx.x] = i2;
        }
      }
      __syncthreads();
    }
    if (threadIdx.x == 0) { chosen[k] = bidx[0]; cval[k] = bval[0]; }
    __syncthreads();
  }

  if (threadIdx.x == 0) {
    float mx = cval[0];
    for (int k = 1; k < TOPKN; ++k) mx = fmaxf(mx, cval[k]);
    float sum = 0.f;
    for (int k = 0; k < TOPKN; ++k) { w[k] = __expf(cval[k] - mx); sum += w[k]; }
    const float inv = 1.f / sum;
    for (int k = 0; k < TOPKN; ++k) w[k] *= inv;
  }
  __syncthreads();

  for (int c = threadIdx.x; c < MEMD; c += blockDim.x) {
    float acc = 0.f;
    for (int k = 0; k < TOPKN; ++k)
      acc += w[k] * memory[(size_t)chosen[k] * MEMD + c];
    mem_lat[(size_t)row * MEMD + c] = acc;
  }
}

// ---------------------------------------------------------------------------
// ConvTranspose2d k=4 s=2 p=1 (+ReLU). w layout [Ci, Co, 4, 4] (torch).
// ---------------------------------------------------------------------------
__global__ void convT_relu(const float* __restrict__ in, const float* __restrict__ w,
                           const float* __restrict__ bias, float* __restrict__ out,
                           int Bn, int Ci, int Co, int Hi, int Ho) {
  const int idx = blockIdx.x * blockDim.x + threadIdx.x;
  const int total = Bn * Co * Ho * Ho;
  if (idx >= total) return;
  const int xx = idx % Ho;
  const int yy = (idx / Ho) % Ho;
  const int o  = (idx / (Ho * Ho)) % Co;
  const int b  = idx / (Ho * Ho * Co);
  float acc = bias[o];
#pragma unroll
  for (int ky = 0; ky < 4; ++ky) {
    const int ny = yy + 1 - ky;
    if (ny & 1) continue;
    const int iy = ny >> 1;
    if (iy < 0 || iy >= Hi) continue;
#pragma unroll
    for (int kx = 0; kx < 4; ++kx) {
      const int nx = xx + 1 - kx;
      if (nx & 1) continue;
      const int ix = nx >> 1;
      if (ix < 0 || ix >= Hi) continue;
      for (int i = 0; i < Ci; ++i)
        acc += in[(((size_t)b * Ci + i) * Hi + iy) * Hi + ix] *
               w[((i * Co + o) * 4 + ky) * 4 + kx];
    }
  }
  out[idx] = fmaxf(acc, 0.f);
}

// Conv2d 3x3 pad 1 (+ReLU). w layout [Co, Ci, 3, 3].
__global__ void conv3x3_relu(const float* __restrict__ in, const float* __restrict__ w,
                             const float* __restrict__ bias, float* __restrict__ out,
                             int Bn, int Ci, int Co, int H) {
  const int idx = blockIdx.x * blockDim.x + threadIdx.x;
  const int total = Bn * Co * H * H;
  if (idx >= total) return;
  const int xx = idx % H;
  const int yy = (idx / H) % H;
  const int o  = (idx / (H * H)) % Co;
  const int b  = idx / (H * H * Co);
  float acc = bias[o];
  for (int i = 0; i < Ci; ++i) {
#pragma unroll
    for (int ky = 0; ky < 3; ++ky) {
      const int iy = yy + ky - 1;
      if (iy < 0 || iy >= H) continue;
#pragma unroll
      for (int kx = 0; kx < 3; ++kx) {
        const int ix = xx + kx - 1;
        if (ix < 0 || ix >= H) continue;
        acc += in[(((size_t)b * Ci + i) * H + iy) * H + ix] *
               w[((o * Ci + i) * 3 + ky) * 3 + kx];
      }
    }
  }
  out[idx] = fmaxf(acc, 0.f);
}

// Final 1x1 conv (Co=1), no activation.
__global__ void conv1x1(const float* __restrict__ in, const float* __restrict__ w,
                        const float* __restrict__ bias, float* __restrict__ out,
                        int Bn, int Ci, int H) {
  const int idx = blockIdx.x * blockDim.x + threadIdx.x;
  const int total = Bn * H * H;
  if (idx >= total) return;
  const int p = idx % (H * H);
  const int b = idx / (H * H);
  float acc = bias[0];
  for (int i = 0; i < Ci; ++i)
    acc += in[((size_t)b * Ci + i) * H * H + p] * w[i];
  out[idx] = acc;
}

// ---------------------------------------------------------------------------
extern "C" void kernel_launch(void* const* d_in, const int* in_sizes, int n_in,
                              void* d_out, int out_size, void* d_ws, size_t ws_size,
                              hipStream_t stream) {
  const float* x        = (const float*)d_in[0];
  const float* patch_w  = (const float*)d_in[1];
  const float* patch_b  = (const float*)d_in[2];
  const float* ln1_g    = (const float*)d_in[3];
  const float* ln1_b    = (const float*)d_in[4];
  const float* inW      = (const float*)d_in[5];
  const float* inB      = (const float*)d_in[6];
  const float* outW     = (const float*)d_in[7];
  const float* outB     = (const float*)d_in[8];
  const float* ln2_g    = (const float*)d_in[9];
  const float* ln2_b    = (const float*)d_in[10];
  const float* f1W      = (const float*)d_in[11];
  const float* f1B      = (const float*)d_in[12];
  const float* f2W      = (const float*)d_in[13];
  const float* f2B      = (const float*)d_in[14];
  const float* lat_w    = (const float*)d_in[15];
  const float* lat_b    = (const float*)d_in[16];
  const float* memory   = (const float*)d_in[17];
  const float* dec_w    = (const float*)d_in[18];
  const float* dec_b    = (const float*)d_in[19];
  const float* upt1_w   = (const float*)d_in[20];
  const float* upt1_b   = (const float*)d_in[21];
  const float* c1_w     = (const float*)d_in[22];
  const float* c1_b     = (const float*)d_in[23];
  const float* upt2_w   = (const float*)d_in[24];
  const float* upt2_b   = (const float*)d_in[25];
  const float* c2_w     = (const float*)d_in[26];
  const float* c2_b     = (const float*)d_in[27];
  const float* c3_w     = (const float*)d_in[28];
  const float* c3_b     = (const float*)d_in[29];
  float* out = (float*)d_out;

  // workspace carve-up (floats)
  float* ws      = (float*)d_ws;
  float* t_flat  = ws;                                // 4096*1568
  float* latent  = t_flat  + (size_t)BATCH * 1568;    // 4096*256
  float* xn      = latent  + (size_t)BATCH * MEMD;
  float* mn      = xn      + (size_t)BATCH * MEMD;
  float* sim     = mn      + (size_t)MEMSZ * MEMD;    // 4096*4096
  float* mem_lat = sim     + (size_t)BATCH * MEMSZ;
  float* dd0     = mem_lat + (size_t)BATCH * MEMD;    // 4096*3136
  float* u1      = dd0     + (size_t)BATCH * 3136;    // 4096*32*14*14
  float* c1o     = u1      + (size_t)BATCH * 32 * 14 * 14;
  float* u2      = c1o     + (size_t)BATCH * 32 * 14 * 14;   // 4096*16*28*28
  float* c2o     = u2      + (size_t)BATCH * 16 * 28 * 28;   // 4096*8*28*28

  const int TPB = 256;

  // 1) fused encoder (patch conv + 2 transformer blocks) -> t_flat
  swin_encoder_kernel<<<BATCH, TPB, 0, stream>>>(
      x, patch_w, patch_b, ln1_g, ln1_b, inW, inB, outW, outB,
      ln2_g, ln2_b, f1W, f1B, f2W, f2B, t_flat);

  // 2) latent = t_flat @ lat_w^T + lat_b   [4096,256], K=1568
  {
    const int blocks = (BATCH / 32) * ((MEMD + 127) / 128);
    gemm_bf16_wmma<<<blocks, TPB, 0, stream>>>(
        t_flat, lat_w, lat_b, latent, BATCH, MEMD, 1568);
  }

  // 3) cosine-normalize latent rows and memory rows
  row_normalize<<<BATCH, TPB, 0, stream>>>(latent, xn, MEMD);
  row_normalize<<<MEMSZ, TPB, 0, stream>>>(memory, mn, MEMD);

  // 4) sim = xn @ mn^T   [4096,4096], K=256
  {
    const int blocks = (BATCH / 32) * ((MEMSZ + 127) / 128);
    gemm_bf16_wmma<<<blocks, TPB, 0, stream>>>(
        xn, mn, nullptr, sim, BATCH, MEMSZ, MEMD);
  }

  // 5) top-10 + softmax + weighted gather of original memory rows
  topk_mem_read<<<BATCH, TPB, 0, stream>>>(sim, memory, mem_lat);

  // 6) dd0 = mem_lat @ dec_w^T + dec_b   [4096,3136], K=256  (== [B,64,7,7])
  {
    const int blocks = (BATCH / 32) * ((3136 + 127) / 128);
    gemm_bf16_wmma<<<blocks, TPB, 0, stream>>>(
        mem_lat, dec_w, dec_b, dd0, BATCH, 3136, MEMD);
  }

  // 7) decoder convs
  {
    int total = BATCH * 32 * 14 * 14;
    convT_relu<<<(total + TPB - 1) / TPB, TPB, 0, stream>>>(
        dd0, upt1_w, upt1_b, u1, BATCH, 64, 32, 7, 14);
  }
  {
    int total = BATCH * 32 * 14 * 14;
    conv3x3_relu<<<(total + TPB - 1) / TPB, TPB, 0, stream>>>(
        u1, c1_w, c1_b, c1o, BATCH, 32, 32, 14);
  }
  {
    int total = BATCH * 16 * 28 * 28;
    convT_relu<<<(total + TPB - 1) / TPB, TPB, 0, stream>>>(
        c1o, upt2_w, upt2_b, u2, BATCH, 32, 16, 14, 28);
  }
  {
    int total = BATCH * 8 * 28 * 28;
    conv3x3_relu<<<(total + TPB - 1) / TPB, TPB, 0, stream>>>(
        u2, c2_w, c2_b, c2o, BATCH, 16, 8, 28);
  }
  {
    int total = BATCH * 28 * 28;
    conv1x1<<<(total + TPB - 1) / TPB, TPB, 0, stream>>>(
        c2o, c3_w, c3_b, out, BATCH, 8, 28);
  }
}
